// Baseline_Jinkyu_21191368638763
// MI455X (gfx1250) — matrix operand order repaired and checked by reading, not compiled, not run on hardware
//
// MI455X (gfx1250, CDNA5) implementation of the conv->attention->LSTM->MLP pipeline.
//
// Strategy: ~63 GFLOP total. All matrix-shaped work runs on
// v_wmma_f32_16x16x32_f16 (f16 in, f32 accumulate). lstm_wih is converted to
// f16 once (63 MB) so the 90 sequential timesteps re-read it from the 192 MB
// L2 instead of HBM. Convs are implicit-GEMM (M=cout tile 16, N=16 output
// pixels per wave, K=cin*kh*kw padded to 32) with im2col tiles staged in LDS.
// proj / LSTM-gates GEMMs load WMMA fragments directly from global memory
// (rows are K-contiguous, 16B aligned). The gates kernel computes two N-tiles
// per wave sharing one A fragment (halves A traffic, doubles WMMA density)
// and prefetches the wih stream (global_prefetch_b8). Elementwise attention /
// softmax / cell update / MLP heads are f32 VALU (negligible FLOPs).
//
// Workspace use: ~231 MB of d_ws (wih_h 63MB, f16 activations, feats/proj f32).

#include <hip/hip_runtime.h>
#include <hip/hip_bf16.h>
#include <math.h>

typedef __attribute__((ext_vector_type(16))) _Float16 v16h;
typedef __attribute__((ext_vector_type(8)))  float    v8f;
typedef __attribute__((ext_vector_type(4)))  unsigned int u32x4;

#define DEV __device__ __forceinline__

// ---------------------------------------------------------------------------
// WMMA fragment loaders (layouts per CDNA5 ISA 7.12.2, 16-bit A 16x32 / B 32x16)
// ---------------------------------------------------------------------------

union FragU { v16h v; u32x4 q[2]; };

// A fragment (16x32, f16), rows row-major with leading dim ldk (halfs).
// lane<16: row=lane,    K = {0..7, 16..23}
// lane>=16: row=lane-16, K = {8..15, 24..31}
DEV v16h frag_a_lds(const _Float16* base, int ldk) {
  int lane = threadIdx.x & 31;
  int r = lane & 15;
  int kb = (lane < 16) ? 0 : 8;
  FragU u;
  u.q[0] = *(const u32x4*)(base + r * ldk + kb);
  u.q[1] = *(const u32x4*)(base + r * ldk + kb + 16);
  return u.v;
}

// A fragment direct from global; rows >= `rows` are zero-padded.
DEV v16h frag_a_g(const _Float16* base, size_t ldk, int kc, int rows) {
  int lane = threadIdx.x & 31;
  int r = lane & 15;
  int kb = (lane < 16) ? 0 : 8;
  FragU u;
  if (r < rows) {
    const _Float16* p = base + (size_t)r * ldk + kc + kb;
    u.q[0] = *(const u32x4*)(p);
    u.q[1] = *(const u32x4*)(p + 16);
  } else {
    u.q[0] = u32x4{0u, 0u, 0u, 0u};
    u.q[1] = u32x4{0u, 0u, 0u, 0u};
  }
  return u.v;
}

// B fragment (32x16, f16). Column n = lane&15 holds K = {0..7, 16..23} for
// lanes 0-15, K = {8..15, 24..31} for lanes 16-31. `base` points at column 0's
// row (row-major by output column, leading dim ldk halfs).
DEV v16h frag_b_lds(const _Float16* base, int ldk) {
  int lane = threadIdx.x & 31;
  int c = lane & 15;
  int kb = (lane < 16) ? 0 : 8;
  FragU u;
  const _Float16* p = base + c * ldk + kb;
  u.q[0] = *(const u32x4*)(p);
  u.q[1] = *(const u32x4*)(p + 16);
  return u.v;
}

DEV v16h frag_b_g(const _Float16* base, size_t ldk, int kc) {
  int lane = threadIdx.x & 31;
  int c = lane & 15;
  int kb = (lane < 16) ? 0 : 8;
  FragU u;
  const _Float16* p = base + (size_t)c * ldk + kc + kb;
  u.q[0] = *(const u32x4*)(p);
  u.q[1] = *(const u32x4*)(p + 16);
  return u.v;
}

DEV v8f wmma32(v16h a, v16h b, v8f c) {
  return __builtin_amdgcn_wmma_f32_16x16x32_f16(false, a, false, b, (short)0, c,
                                                false, false);
}

// ---------------------------------------------------------------------------
// f32 -> f16 conversion (weights done once per launch; reused 90x by scan)
// ---------------------------------------------------------------------------
__global__ void f32_to_f16_kernel(const float* __restrict__ src,
                                  _Float16* __restrict__ dst, long n) {
  long i = (long)blockIdx.x * blockDim.x + threadIdx.x;
  long stride = (long)gridDim.x * blockDim.x;
  for (; i < n; i += stride) dst[i] = (_Float16)src[i];
}

// ---------------------------------------------------------------------------
// Implicit-GEMM convolution + ReLU via WMMA.
//   out[co, pix] = relu( sum_k W[co,k] * im2col[k,pix] + bias[co] )
// Block: 128 threads (4 waves). Wave w computes a 16(co) x 16(pix) tile.
// gridDim = { ceil(NPIX/64), ceil(COUT/16), 360 frames }.
// WRITE_FEATS: last conv writes the torch-permute feats layout [frame][L][D]
//   (both f32 for elementwise attention and f16 for the proj GEMM).
// ---------------------------------------------------------------------------
template <int CIN, int COUT, int KK, int STRIDE, int PAD, int HIN, int WIN,
          int HOUT, int WOUT, bool IN_F32, bool WRITE_FEATS>
__global__ void conv_wmma_kernel(const void* __restrict__ in_,
                                 const float* __restrict__ w,
                                 const float* __restrict__ bias,
                                 _Float16* __restrict__ out,
                                 float* __restrict__ feats,
                                 _Float16* __restrict__ featsh) {
  constexpr int KREAL = CIN * KK * KK;
  constexpr int KPAD = ((KREAL + 31) / 32) * 32;
  constexpr int NPIX = HOUT * WOUT;

  __shared__ _Float16 Atile[16 * KPAD];   // weights, f16, zero-padded
  __shared__ _Float16 Bt[4][16 * 32];     // per-wave im2col chunk

  const int frame = blockIdx.z;
  const int coTile = blockIdx.y;
  const int tid = threadIdx.x;
  const int lane = tid & 31;
  const int wv = tid >> 5;
  const int pixbase = (blockIdx.x * 4 + wv) * 16;

  // Stage (and f16-convert) this block's 16 output-channel rows of weights.
  for (int idx = tid; idx < 16 * KPAD; idx += blockDim.x) {
    int cl = idx / KPAD, k = idx % KPAD;
    int co = coTile * 16 + cl;
    float v = (co < COUT && k < KREAL) ? w[(size_t)co * KREAL + k] : 0.f;
    Atile[cl * KPAD + k] = (_Float16)v;
  }
  __syncthreads();

  const float* inF = (const float*)in_;
  const _Float16* inH = (const _Float16*)in_;
  const size_t inFrame = (size_t)frame * CIN * HIN * WIN;

  v8f acc = {0.f, 0.f, 0.f, 0.f, 0.f, 0.f, 0.f, 0.f};

  for (int kc = 0; kc < KPAD; kc += 32) {
    // im2col gather: this wave fills its 16(pix) x 32(k) tile; each lane owns
    // one k column. LDS ops within a wave are in-order (DScnt), no barrier.
    int k = kc + lane;
    int ci = k / (KK * KK);
    int rem = k % (KK * KK);
    int kh = rem / KK, kw = rem % KK;
    bool kvalid = (ci < CIN);
    for (int p = 0; p < 16; ++p) {
      int pix = pixbase + p;
      float v = 0.f;
      if (kvalid && pix < NPIX) {
        int oy = pix / WOUT, ox = pix % WOUT;
        int iy = oy * STRIDE - PAD + kh;
        int ix = ox * STRIDE - PAD + kw;
        if (iy >= 0 && iy < HIN && ix >= 0 && ix < WIN) {
          size_t off = inFrame + ((size_t)ci * HIN + iy) * WIN + ix;
          v = IN_F32 ? inF[off] : (float)inH[off];
        }
      }
      Bt[wv][p * 32 + lane] = (_Float16)v;
    }
    v16h a = frag_a_lds(Atile + kc, KPAD);
    v16h b = frag_b_lds(Bt[wv], 32);
    acc = wmma32(a, b, acc);
  }

  // Store. C/D layout: vgpr r, lane l -> M = r + 8*(l/16), N = l%16.
  int nl = lane & 15, hi = lane >> 4;
  int pix = pixbase + nl;
  if (pix < NPIX) {
    int oy = pix / WOUT, ox = pix % WOUT;
    for (int r = 0; r < 8; ++r) {
      int co = coTile * 16 + r + 8 * hi;
      if (co < COUT) {
        float v = fmaxf(acc[r] + bias[co], 0.f);
        if (WRITE_FEATS) {
          // torch: permute(0,3,1,2).view(-1,64,240).permute(0,2,1)
          int flat = ox * (64 * 12) + co * 12 + oy;  // (w, c, h) flatten
          int d = flat / 240, l = flat % 240;
          size_t o = (size_t)frame * 15360 + l * 64 + d;
          feats[o] = v;
          featsh[o] = (_Float16)v;
        } else {
          out[((size_t)frame * COUT + co) * NPIX + pix] = (_Float16)v;
        }
      }
    }
  }
}

// ---------------------------------------------------------------------------
// proj = feats @ proj_w.T + proj_b    (86400 x 64 x 64), WMMA, frags from global
// grid.x = 86400/16 row tiles; block 128 (wave w owns N tile w*16).
// ---------------------------------------------------------------------------
__global__ void proj_wmma_kernel(const _Float16* __restrict__ featsh,
                                 const _Float16* __restrict__ projw_h,
                                 const float* __restrict__ proj_b,
                                 float* __restrict__ proj) {
  const int lane = threadIdx.x & 31;
  const int wv = threadIdx.x >> 5;
  const int row0 = blockIdx.x * 16;
  const int n0 = wv * 16;

  v8f acc = {0.f, 0.f, 0.f, 0.f, 0.f, 0.f, 0.f, 0.f};
  for (int kc = 0; kc < 64; kc += 32) {
    v16h a = frag_a_g(featsh + (size_t)row0 * 64, 64, kc, 16);
    v16h b = frag_b_g(projw_h + (size_t)n0 * 64, 64, kc);  // B[n][k]=proj_w[n][k]
    acc = wmma32(a, b, acc);
  }
  int nl = lane & 15, hi = lane >> 4;
  int n = n0 + nl;
  float bn = proj_b[n];
  for (int r = 0; r < 8; ++r) {
    int row = row0 + r + 8 * hi;
    proj[(size_t)row * 64 + n] = acc[r] + bn;
  }
}

// ---------------------------------------------------------------------------
// Per-timestep attention + context (tiny; f32 VALU). One block per batch b.
// ---------------------------------------------------------------------------
__global__ void attn_context_kernel(const float* __restrict__ feats,
                                    const float* __restrict__ proj,
                                    const float* __restrict__ hx,
                                    const float* __restrict__ w_w,   // [64][512]
                                    const float* __restrict__ w_b,   // [64]
                                    const float* __restrict__ wattn_w,  // [64]
                                    const float* __restrict__ wattn_b,  // [1]
                                    _Float16* __restrict__ ctx_h, int t) {
  const int b = blockIdx.x;
  const int tid = threadIdx.x;
  const size_t frame = (size_t)b * 90 + t;
  const float* fp = proj + frame * 15360;
  const float* ff = feats + frame * 15360;

  __shared__ float wh[64];
  __shared__ float sc[256];
  __shared__ float red[256];

  if (tid < 64) {
    float s = w_b[tid];
    const float* wr = w_w + tid * 512;
    const float* h = hx + b * 512;
    for (int j = 0; j < 512; ++j) s += h[j] * wr[j];
    wh[tid] = s;
  }
  __syncthreads();

  if (tid < 240) {
    float s = 0.f;
    const float* pr = fp + tid * 64;
    for (int d = 0; d < 64; ++d) s += tanhf(pr[d] + wh[d]) * wattn_w[d];
    sc[tid] = s + wattn_b[0];
  }
  __syncthreads();

  // softmax over 240 locations
  red[tid] = (tid < 240) ? sc[tid] : -1e30f;
  __syncthreads();
  for (int s = 128; s > 0; s >>= 1) {
    if (tid < s) red[tid] = fmaxf(red[tid], red[tid + s]);
    __syncthreads();
  }
  float mx = red[0];
  __syncthreads();
  float e = (tid < 240) ? __expf(sc[tid] - mx) : 0.f;
  red[tid] = e;
  __syncthreads();
  for (int s = 128; s > 0; s >>= 1) {
    if (tid < s) red[tid] += red[tid + s];
    __syncthreads();
  }
  float inv = 1.f / red[0];
  __syncthreads();
  if (tid < 240) sc[tid] = e * inv;  // alpha
  __syncthreads();

  // context[b, l*64+d] = feats[l,d] * alpha[l], stored f16 for the WMMA GEMM
  for (int idx = tid; idx < 15360; idx += blockDim.x) {
    int l = idx >> 6;
    ctx_h[(size_t)b * 15360 + idx] = (_Float16)(ff[idx] * sc[l]);
  }
}

// ---------------------------------------------------------------------------
// gates = ctx @ wih.T + hx @ whh.T + bih + bhh   (M=4 padded to 16, N=2048)
// Critical sequential-path GEMM (runs 90x). Each wave owns TWO adjacent
// 16-col N-tiles sharing one A fragment: per K-chunk = 2 A-loads + 4 B-loads
// + 2 WMMA. wih_h (63 MB f16) is L2-resident after t=0; global_prefetch_b8
// pulls the next 4KB of each wave's B rows toward the WGP.
// grid.x = 16 (N tiles of 128), block 128 (4 waves).
// ---------------------------------------------------------------------------
__global__ void lstm_gates_wmma_kernel(const _Float16* __restrict__ ctx_h,
                                       const _Float16* __restrict__ hx_h,
                                       const _Float16* __restrict__ wih_h,
                                       const _Float16* __restrict__ whh_h,
                                       const float* __restrict__ bih,
                                       const float* __restrict__ bhh,
                                       float* __restrict__ gates) {
  const int lane = threadIdx.x & 31;
  const int wv = threadIdx.x >> 5;
  const int n0 = (blockIdx.x * 4 + wv) * 32;   // two tiles: n0, n0+16
  const int col = lane & 15;

  const _Float16* brow0 = wih_h + (size_t)n0 * 15360;
  const _Float16* brow1 = wih_h + (size_t)(n0 + 16) * 15360;

  v8f acc0 = {0.f, 0.f, 0.f, 0.f, 0.f, 0.f, 0.f, 0.f};
  v8f acc1 = {0.f, 0.f, 0.f, 0.f, 0.f, 0.f, 0.f, 0.f};

#pragma unroll 4
  for (int kc = 0; kc < 15360; kc += 32) {           // 480 chunks, 960 WMMA/wave
    // prefetch this lane's B rows ~4KB ahead (L2 -> WGP)
    if (kc + 2048 < 15360) {
      __builtin_prefetch(brow0 + (size_t)col * 15360 + kc + 2048, 0, 1);
      __builtin_prefetch(brow1 + (size_t)col * 15360 + kc + 2048, 0, 1);
    }
    v16h a = frag_a_g(ctx_h, 15360, kc, 4);
    v16h b0 = frag_b_g(brow0, 15360, kc);
    v16h b1 = frag_b_g(brow1, 15360, kc);
    acc0 = wmma32(a, b0, acc0);
    acc1 = wmma32(a, b1, acc1);
  }
#pragma unroll 4
  for (int kc = 0; kc < 512; kc += 32) {             // recurrent term
    v16h a = frag_a_g(hx_h, 512, kc, 4);
    v16h b0 = frag_b_g(whh_h + (size_t)n0 * 512, 512, kc);
    v16h b1 = frag_b_g(whh_h + (size_t)(n0 + 16) * 512, 512, kc);
    acc0 = wmma32(a, b0, acc0);
    acc1 = wmma32(a, b1, acc1);
  }
  if (lane < 16) {  // rows 0..3 are the real batch rows (M pad elsewhere)
    int na = n0 + col, nb = n0 + 16 + col;
    float ba = bih[na] + bhh[na];
    float bb = bih[nb] + bhh[nb];
    for (int r = 0; r < 4; ++r) {
      gates[(size_t)r * 2048 + na] = acc0[r] + ba;
      gates[(size_t)r * 2048 + nb] = acc1[r] + bb;
    }
  }
}

// ---------------------------------------------------------------------------
// LSTM cell update (torch gate order i,f,g,o). 2048 threads = (b, j).
// ---------------------------------------------------------------------------
DEV float sigm(float x) { return 1.f / (1.f + __expf(-x)); }

__global__ void lstm_update_kernel(const float* __restrict__ gates,
                                   float* __restrict__ hx, float* __restrict__ cx,
                                   _Float16* __restrict__ hx_h) {
  int id = blockIdx.x * blockDim.x + threadIdx.x;  // 0..2047
  if (id >= 2048) return;
  int b = id >> 9, j = id & 511;
  const float* g = gates + (size_t)b * 2048;
  float gi = g[j], gf = g[512 + j], gg = g[1024 + j], go = g[1536 + j];
  float c = sigm(gf) * cx[id] + sigm(gi) * tanhf(gg);
  cx[id] = c;
  float h = sigm(go) * tanhf(c);
  hx[id] = h;
  hx_h[id] = (_Float16)h;
}

__global__ void zero_state_kernel(float* hx, float* cx, _Float16* hx_h) {
  int id = blockIdx.x * blockDim.x + threadIdx.x;
  if (id < 2048) { hx[id] = 0.f; cx[id] = 0.f; hx_h[id] = (_Float16)0.f; }
}

// ---------------------------------------------------------------------------
// steer / vel MLP heads (512->100->50->10->1). grid = {4 batch, 2 head}.
// ---------------------------------------------------------------------------
__global__ void heads_kernel(const float* __restrict__ hx,
    const float* sw1, const float* sb1, const float* sw2, const float* sb2,
    const float* sw3, const float* sb3, const float* sw4, const float* sb4,
    const float* vw1, const float* vb1, const float* vw2, const float* vb2,
    const float* vw3, const float* vb3, const float* vw4, const float* vb4,
    float* __restrict__ out, int t) {
  const int b = blockIdx.x;
  const int head = blockIdx.y;
  const float* w1 = head ? vw1 : sw1; const float* b1 = head ? vb1 : sb1;
  const float* w2 = head ? vw2 : sw2; const float* b2 = head ? vb2 : sb2;
  const float* w3 = head ? vw3 : sw3; const float* b3 = head ? vb3 : sb3;
  const float* w4 = head ? vw4 : sw4; const float* b4 = head ? vb4 : sb4;
  const float* h = hx + b * 512;
  const int tid = threadIdx.x;

  __shared__ float h1[100], h2[50], h3[10];
  for (int o = tid; o < 100; o += blockDim.x) {
    float s = b1[o];
    const float* wr = w1 + o * 512;
    for (int j = 0; j < 512; ++j) s += h[j] * wr[j];
    h1[o] = fmaxf(s, 0.f);
  }
  __syncthreads();
  for (int o = tid; o < 50; o += blockDim.x) {
    float s = b2[o];
    const float* wr = w2 + o * 100;
    for (int j = 0; j < 100; ++j) s += h1[j] * wr[j];
    h2[o] = fmaxf(s, 0.f);
  }
  __syncthreads();
  for (int o = tid; o < 10; o += blockDim.x) {
    float s = b3[o];
    const float* wr = w3 + o * 50;
    for (int j = 0; j < 50; ++j) s += h2[j] * wr[j];
    h3[o] = fmaxf(s, 0.f);
  }
  __syncthreads();
  if (tid == 0) {
    float s = b4[0];
    for (int j = 0; j < 10; ++j) s += h3[j] * w4[j];
    out[head * 360 + t * 4 + b] = s;  // steers at [0,360), vels at [360,720)
  }
}

// ---------------------------------------------------------------------------
// Host launcher
// ---------------------------------------------------------------------------
extern "C" void kernel_launch(void* const* d_in, const int* in_sizes, int n_in,
                              void* d_out, int out_size, void* d_ws,
                              size_t ws_size, hipStream_t stream) {
  (void)in_sizes; (void)n_in; (void)out_size; (void)ws_size;

  const float* cam = (const float*)d_in[0];
  const float* cw[5] = {(const float*)d_in[1], (const float*)d_in[3],
                        (const float*)d_in[5], (const float*)d_in[7],
                        (const float*)d_in[9]};
  const float* cb[5] = {(const float*)d_in[2], (const float*)d_in[4],
                        (const float*)d_in[6], (const float*)d_in[8],
                        (const float*)d_in[10]};
  const float* proj_w = (const float*)d_in[11];
  const float* proj_b = (const float*)d_in[12];
  const float* w_w = (const float*)d_in[13];
  const float* w_b = (const float*)d_in[14];
  const float* wattn_w = (const float*)d_in[15];
  const float* wattn_b = (const float*)d_in[16];
  const float* lstm_wih = (const float*)d_in[17];
  const float* lstm_whh = (const float*)d_in[18];
  const float* lstm_bih = (const float*)d_in[19];
  const float* lstm_bhh = (const float*)d_in[20];
  const float* sw[8]; const float* vw[8];
  for (int i = 0; i < 8; ++i) { sw[i] = (const float*)d_in[21 + i]; vw[i] = (const float*)d_in[29 + i]; }
  float* out = (float*)d_out;

  // ---- workspace carve (256B aligned) ----
  char* p = (char*)d_ws;
  auto alloc = [&](size_t bytes) -> char* {
    char* r = p; p += (bytes + 255) & ~(size_t)255; return r;
  };
  const long N_WIH = 2048L * 15360, N_WHH = 2048L * 512, N_PW = 64L * 64;
  _Float16* wih_h  = (_Float16*)alloc(N_WIH * 2);
  _Float16* whh_h  = (_Float16*)alloc(N_WHH * 2);
  _Float16* projw_h = (_Float16*)alloc(N_PW * 2);
  _Float16* act1 = (_Float16*)alloc(360L * 24 * 48 * 80 * 2);
  _Float16* act2 = (_Float16*)alloc(360L * 36 * 24 * 40 * 2);
  _Float16* act3 = (_Float16*)alloc(360L * 48 * 240 * 2);
  _Float16* act4 = (_Float16*)alloc(360L * 64 * 240 * 2);
  float*    feats  = (float*)alloc(360L * 15360 * 4);
  _Float16* featsh = (_Float16*)alloc(360L * 15360 * 2);
  float*    proj   = (float*)alloc(360L * 15360 * 4);
  _Float16* ctx_h  = (_Float16*)alloc(4L * 15360 * 2);
  float*    gates  = (float*)alloc(4L * 2048 * 4);
  float*    hx     = (float*)alloc(2048 * 4);
  float*    cx     = (float*)alloc(2048 * 4);
  _Float16* hx_h   = (_Float16*)alloc(2048 * 2);

  // ---- one-time (per launch) weight conversions to f16 ----
  f32_to_f16_kernel<<<8192, 256, 0, stream>>>(lstm_wih, wih_h, N_WIH);
  f32_to_f16_kernel<<<2048, 256, 0, stream>>>(lstm_whh, whh_h, N_WHH);
  f32_to_f16_kernel<<<16, 256, 0, stream>>>(proj_w, projw_h, N_PW);
  zero_state_kernel<<<8, 256, 0, stream>>>(hx, cx, hx_h);

  // ---- conv stack (implicit GEMM WMMA), parallel over 360 frames ----
  conv_wmma_kernel<3, 24, 5, 2, 2, 96, 160, 48, 80, true, false>
      <<<dim3(60, 2, 360), 128, 0, stream>>>(cam, cw[0], cb[0], act1, nullptr, nullptr);
  conv_wmma_kernel<24, 36, 5, 2, 2, 48, 80, 24, 40, false, false>
      <<<dim3(15, 3, 360), 128, 0, stream>>>(act1, cw[1], cb[1], act2, nullptr, nullptr);
  conv_wmma_kernel<36, 48, 5, 2, 2, 24, 40, 12, 20, false, false>
      <<<dim3(4, 3, 360), 128, 0, stream>>>(act2, cw[2], cb[2], act3, nullptr, nullptr);
  conv_wmma_kernel<48, 64, 3, 1, 1, 12, 20, 12, 20, false, false>
      <<<dim3(4, 4, 360), 128, 0, stream>>>(act3, cw[3], cb[3], act4, nullptr, nullptr);
  conv_wmma_kernel<64, 64, 3, 1, 1, 12, 20, 12, 20, false, true>
      <<<dim3(4, 4, 360), 128, 0, stream>>>(act4, cw[4], cb[4], nullptr, feats, featsh);

  // ---- projection GEMM over all (frame, location) rows ----
  proj_wmma_kernel<<<dim3(86400 / 16), 128, 0, stream>>>(featsh, projw_h, proj_b, proj);

  // ---- sequential scan over T=90 (graph replay amortizes launch cost) ----
  for (int t = 0; t < 90; ++t) {
    attn_context_kernel<<<dim3(4), 256, 0, stream>>>(feats, proj, hx, w_w, w_b,
                                                     wattn_w, wattn_b, ctx_h, t);
    lstm_gates_wmma_kernel<<<dim3(16), 128, 0, stream>>>(ctx_h, hx_h, wih_h,
                                                         whh_h, lstm_bih,
                                                         lstm_bhh, gates);
    lstm_update_kernel<<<dim3(8), 256, 0, stream>>>(gates, hx, cx, hx_h);
    heads_kernel<<<dim3(4, 2), 128, 0, stream>>>(hx,
        sw[0], sw[1], sw[2], sw[3], sw[4], sw[5], sw[6], sw[7],
        vw[0], vw[1], vw[2], vw[3], vw[4], vw[5], vw[6], vw[7], out, t);
  }
}
